// INRLoe_11416023072850
// MI455X (gfx1250) — matrix operand order repaired and checked by reading, not compiled, not run on hardware
//
#include <hip/hip_runtime.h>

// ---------------------------------------------------------------------------
// INRLoe forward for MI455X (gfx1250, wave32, WMMA).
// Middle layers: v_wmma_f32_16x16x32_bf16 with hi/lo bf16 splitting of both
// activations and weights (3 WMMAs per 32-K chunk) -> near-fp32 accuracy at
// bf16 matrix-pipe rates. Weight splitting is hoisted into a bandwidth-bound
// pre-pass that emits fragments pre-swizzled in WMMA B layout, so the GEMM
// inner loop is pure b128-loads + WMMA (no VALU conversion wall).
// ---------------------------------------------------------------------------

typedef __attribute__((ext_vector_type(16))) __bf16 v16bf;
typedef __attribute__((ext_vector_type(8)))  float  v8f;

union FragAB { v16bf v; unsigned short u[16]; uint4 q[2]; };
union FragC  { v8f   v; float f[8]; };

__device__ __forceinline__ unsigned short f2bf(float x) {
    unsigned int u = __float_as_uint(x);
    unsigned int r = u + 0x7FFFu + ((u >> 16) & 1u);   // round-to-nearest-even
    return (unsigned short)(r >> 16);
}
__device__ __forceinline__ float bf2f(unsigned short h) {
    return __uint_as_float(((unsigned int)h) << 16);
}

#define GATE_STRIDE 1368   // sum(NUM_EXPS)
#define NC 1024
#define HID 256

// ---------------------------------------------------------------------------
// Kernel 1: top-k gate compaction. grid=(5 groups, 2 imgs), block=256.
// rank(e) = #{j : |v_j|>|v_e|} + #{j<e : |v_j|==|v_e|}  (matches lax.top_k
// tie-breaking); keep iff rank < K. Slot = deterministic prefix count.
// ---------------------------------------------------------------------------
__global__ __launch_bounds__(256) void gate_topk(const float* __restrict__ raw,
                                                 int* __restrict__ selIdx,
                                                 float* __restrict__ selVal) {
    const int offs[5] = {0, 8, 24, 88, 344};
    const int Es[5]   = {8, 16, 64, 256, 1024};
    const int Kg[5]   = {4, 4, 32, 32, 256};
    const int base[5] = {0, 8, 16, 80, 144};   // int offsets into sel arrays

    const int g = blockIdx.x, img = blockIdx.y, t = threadIdx.x;
    const int E = Es[g], K = Kg[g];
    const float* src = raw + img * GATE_STRIDE + offs[g];

    __shared__ float sa[1024];
    __shared__ int   keep[1024];

    for (int e = t; e < E; e += 256) sa[e] = fabsf(src[e]);
    __syncthreads();
    for (int e = t; e < E; e += 256) {
        const float a = sa[e];
        int cnt = 0;
        for (int j = 0; j < E; ++j) {
            const float bx = sa[j];
            cnt += (bx > a) || (bx == a && j < e);
        }
        keep[e] = (cnt < K) ? 1 : 0;
    }
    __syncthreads();
    for (int e = t; e < E; e += 256) {
        if (keep[e]) {
            int slot = 0;
            for (int j = 0; j < e; ++j) slot += keep[j];
            const int o = base[g] + img * K + slot;
            selIdx[o] = e;
            selVal[o] = src[e];
        }
    }
}

// ---------------------------------------------------------------------------
// Kernel 2: layer 0.  x0[i,c,d] = sum_s g * sin(30*(coords[c]·W0[d*8+e] + b0))
// Writes hi/lo bf16 activations. grid=1024, block=256.
// ---------------------------------------------------------------------------
__global__ __launch_bounds__(256) void layer0_kernel(
        const float* __restrict__ coords, const float* __restrict__ W0,
        const float* __restrict__ b0, const int* __restrict__ selIdx,
        const float* __restrict__ selVal,
        unsigned short* __restrict__ yhi, unsigned short* __restrict__ ylo) {
    const int idx = blockIdx.x * 256 + threadIdx.x;   // over NC*HID
    const int c = idx >> 8, d = idx & 255;
    const float cx = coords[c * 2 + 0], cy = coords[c * 2 + 1];
    for (int img = 0; img < 2; ++img) {
        float acc = 0.f;
        #pragma unroll
        for (int s = 0; s < 4; ++s) {
            const int   e = selIdx[img * 4 + s];
            const float g = selVal[img * 4 + s];
            const int   r = d * 8 + e;
            const float t = cx * W0[r * 2 + 0] + cy * W0[r * 2 + 1] + b0[r];
            acc += g * __sinf(30.f * t);
        }
        const size_t o = ((size_t)(img * NC + c)) * HID + d;
        const unsigned short h = f2bf(acc);
        yhi[o] = h;
        ylo[o] = f2bf(acc - bf2f(h));
    }
}

// ---------------------------------------------------------------------------
// Kernel 3a: weight pre-pass. Gathers selected expert slices, splits fp32 ->
// bf16 hi/lo, and stores pre-swizzled in WMMA B-fragment order:
//   index = (((img*Ksel + s)*16 + nb)*8 + ch)*32 + lane, 16 bf16 each, where
//   element j of a lane is W[(d*E+e)*256 + k], d = nb*16 + (lane&15),
//   k = ch*32 + (lane>>4)*16 + j.
// One thread per fragment-lane (16 values). Reads 64B/thread, writes 2x32B.
// ---------------------------------------------------------------------------
__global__ __launch_bounds__(256) void prep_weights(
        const float* __restrict__ W, const int* __restrict__ selIdx,
        int Ksel, int E,
        unsigned short* __restrict__ whi, unsigned short* __restrict__ wlo) {
    const int gid = blockIdx.x * 256 + threadIdx.x;
    const int total = 2 * Ksel * 4096;        // img * slot * nb(16)*ch(8)*lane(32)
    if (gid >= total) return;
    const int lane = gid & 31;
    const int ch   = (gid >> 5) & 7;
    const int nb   = (gid >> 8) & 15;
    const int s    = (gid >> 12) % Ksel;
    const int img  = gid / (Ksel << 12);

    const int e  = selIdx[img * Ksel + s];
    const int d  = nb * 16 + (lane & 15);
    const int kb = ch * 32 + (lane >> 4) * 16;
    const float* wp = W + ((size_t)d * E + e) * HID + kb;

    unsigned short hu[16], lu[16];
    #pragma unroll
    for (int q = 0; q < 4; ++q) {
        const float4 t4 = ((const float4*)wp)[q];
        const float v[4] = {t4.x, t4.y, t4.z, t4.w};
        #pragma unroll
        for (int z = 0; z < 4; ++z) {
            const unsigned short h = f2bf(v[z]);
            hu[4 * q + z] = h;
            lu[4 * q + z] = f2bf(v[z] - bf2f(h));
        }
    }
    uint4* oh = (uint4*)(whi + (size_t)gid * 16);
    uint4* ol = (uint4*)(wlo + (size_t)gid * 16);
    oh[0] = ((const uint4*)hu)[0]; oh[1] = ((const uint4*)hu)[1];
    ol[0] = ((const uint4*)lu)[0]; ol[1] = ((const uint4*)lu)[1];
}

// ---------------------------------------------------------------------------
// Kernel 3b: middle MoE layer GEMM (layers 1..3).
// grid = (NC/32, HID/64, 2 imgs), block = 256 (8 waves).
// Wave tile 16x16; block tile 32(M coords) x 64(N dims).
// A (activations, hi/lo bf16) preloaded in VGPRs across the expert loop;
// B fragments come pre-swizzled from prep_weights: 4 coalesced b128 loads +
// 3 x v_wmma_f32_16x16x32_bf16 per 32-K chunk (Ah*Bh + Ah*Bl + Al*Bh).
// ---------------------------------------------------------------------------
__global__ __launch_bounds__(256) void moe_layer(
        const unsigned short* __restrict__ whi, const unsigned short* __restrict__ wlo,
        const float* __restrict__ bias,
        const unsigned short* __restrict__ xhi, const unsigned short* __restrict__ xlo,
        const int* __restrict__ selIdx, const float* __restrict__ selVal,
        int Ksel, int E,
        unsigned short* __restrict__ yhi, unsigned short* __restrict__ ylo) {
    const int tid    = threadIdx.x;
    const int lane   = tid & 31;
    const int wave   = tid >> 5;
    const int laneHi = lane >> 4;       // 0: lanes 0-15, 1: lanes 16-31
    const int lane15 = lane & 15;
    const int img    = blockIdx.z;
    const int mTile  = blockIdx.x * 32 + (wave & 1) * 16;
    const int nTile  = blockIdx.y * 64 + (wave >> 1) * 16;
    const int nb     = nTile >> 4;      // B-fragment n-block index

    // --- Preload A fragments for full K=256 (8 chunks, hi & lo) -------------
    // A k order chosen here: lanes0-15 hold K=0..15, lanes16-31 K=16..31 (matches B pre-swizzle)
    const int aRow = mTile + lane15;
    const unsigned short* axh = xhi + ((size_t)(img * NC + aRow)) * HID;
    const unsigned short* axl = xlo + ((size_t)(img * NC + aRow)) * HID;
    FragAB Ah[8], Al[8];
    #pragma unroll
    for (int ch = 0; ch < 8; ++ch) {
        const int kb = ch * 32 + laneHi * 16;
        Ah[ch].q[0] = *(const uint4*)(axh + kb);
        Ah[ch].q[1] = *(const uint4*)(axh + kb + 8);
        Al[ch].q[0] = *(const uint4*)(axl + kb);
        Al[ch].q[1] = *(const uint4*)(axl + kb + 8);
    }

    float acc[8];
    #pragma unroll
    for (int j = 0; j < 8; ++j) acc[j] = 0.f;

    const int dCol = nTile + lane15;    // output dim handled by this lane (bias & C)

    for (int s = 0; s < Ksel; ++s) {
        const int   e = selIdx[img * Ksel + s];
        const float g = selVal[img * Ksel + s];

        FragC c;
        #pragma unroll
        for (int j = 0; j < 8; ++j) c.f[j] = 0.f;

        #pragma unroll
        for (int ch = 0; ch < 8; ++ch) {
            const size_t fo =
                ((((size_t)((img * Ksel + s) * 16 + nb) * 8 + ch) * 32) + lane) * 16;
            FragAB Bh, Bl;
            Bh.q[0] = *(const uint4*)(whi + fo);
            Bh.q[1] = *(const uint4*)(whi + fo + 8);
            Bl.q[0] = *(const uint4*)(wlo + fo);
            Bl.q[1] = *(const uint4*)(wlo + fo + 8);
            c.v = __builtin_amdgcn_wmma_f32_16x16x32_bf16(
                      false, Ah[ch].v, false, Bh.v, (short)0, c.v, false, false);
            c.v = __builtin_amdgcn_wmma_f32_16x16x32_bf16(
                      false, Ah[ch].v, false, Bl.v, (short)0, c.v, false, false);
            c.v = __builtin_amdgcn_wmma_f32_16x16x32_bf16(
                      false, Al[ch].v, false, Bh.v, (short)0, c.v, false, false);
        }
        const float bv = bias[(size_t)dCol * E + e];
        #pragma unroll
        for (int j = 0; j < 8; ++j)
            acc[j] += g * __sinf(30.f * (c.f[j] + bv));
    }

    // C layout: lanes0-15 -> M=j, lanes16-31 -> M=8+j; N = lane&15.
    #pragma unroll
    for (int j = 0; j < 8; ++j) {
        const int m = mTile + laneHi * 8 + j;
        const size_t o = ((size_t)(img * NC + m)) * HID + dCol;
        const float v = acc[j];
        const unsigned short h = f2bf(v);
        yhi[o] = h;
        ylo[o] = f2bf(v - bf2f(h));
    }
}

// ---------------------------------------------------------------------------
// Kernel 4: last layer is linear -> pre-combine 256 selected experts:
// Weff[i,d,k] = sum_s g*W4[(d*1024+e)*256+k]; beff likewise.
// grid=(3,2), block=256 (thread = k).
// ---------------------------------------------------------------------------
__global__ __launch_bounds__(256) void combine4_kernel(
        const float* __restrict__ W4, const float* __restrict__ b4,
        const int* __restrict__ selIdx, const float* __restrict__ selVal,
        float* __restrict__ Weff, float* __restrict__ beff) {
    const int d = blockIdx.x, img = blockIdx.y, k = threadIdx.x;
    const int*   si = selIdx + img * 256;
    const float* sv = selVal + img * 256;
    float acc = 0.f;
    for (int s = 0; s < 256; ++s) {
        const int e = si[s];
        acc += sv[s] * W4[((size_t)d * 1024 + e) * HID + k];
    }
    Weff[((size_t)(img * 3 + d)) * HID + k] = acc;
    if (k == 0) {
        float ba = 0.f;
        for (int s = 0; s < 256; ++s) ba += sv[s] * b4[d * 1024 + si[s]];
        beff[img * 3 + d] = ba;
    }
}

// ---------------------------------------------------------------------------
// Kernel 5: out[i,c,:] = x3[i,c,:] @ Weff[i]^T + beff[i].  grid=8, block=256.
// ---------------------------------------------------------------------------
__global__ __launch_bounds__(256) void final4_kernel(
        const unsigned short* __restrict__ xhi, const unsigned short* __restrict__ xlo,
        const float* __restrict__ Weff, const float* __restrict__ beff,
        float* __restrict__ out) {
    const int gidx = blockIdx.x * 256 + threadIdx.x;   // over 2*NC
    const int img = gidx >> 10, c = gidx & (NC - 1);
    float a0 = beff[img * 3 + 0], a1 = beff[img * 3 + 1], a2 = beff[img * 3 + 2];
    const size_t xo = ((size_t)(img * NC + c)) * HID;
    const float* w0 = Weff + ((size_t)(img * 3 + 0)) * HID;
    const float* w1 = Weff + ((size_t)(img * 3 + 1)) * HID;
    const float* w2 = Weff + ((size_t)(img * 3 + 2)) * HID;
    for (int k = 0; k < HID; ++k) {
        const float xv = bf2f(xhi[xo + k]) + bf2f(xlo[xo + k]);
        a0 = fmaf(xv, w0[k], a0);
        a1 = fmaf(xv, w1[k], a1);
        a2 = fmaf(xv, w2[k], a2);
    }
    float* o = out + ((size_t)(img * NC + c)) * 3;
    o[0] = a0; o[1] = a1; o[2] = a2;
}

// ---------------------------------------------------------------------------
extern "C" void kernel_launch(void* const* d_in, const int* in_sizes, int n_in,
                              void* d_out, int out_size, void* d_ws, size_t ws_size,
                              hipStream_t stream) {
    const float* raw    = (const float*)d_in[0];
    const float* coords = (const float*)d_in[1];
    const float* W0 = (const float*)d_in[2];  const float* b0 = (const float*)d_in[3];
    const float* W1 = (const float*)d_in[4];  const float* b1 = (const float*)d_in[5];
    const float* W2 = (const float*)d_in[6];  const float* b2 = (const float*)d_in[7];
    const float* W3 = (const float*)d_in[8];  const float* b3 = (const float*)d_in[9];
    const float* W4 = (const float*)d_in[10]; const float* b4 = (const float*)d_in[11];

    // Workspace layout (~20.1 MB):
    char* ws = (char*)d_ws;
    int*   selIdx = (int*)ws;                        // 656 ints
    float* selVal = (float*)(ws + 4096);             // 656 floats
    const size_t BUF = (size_t)2 * NC * HID * sizeof(unsigned short);  // 1 MB
    unsigned short* Ahi = (unsigned short*)(ws + 8192);
    unsigned short* Alo = (unsigned short*)(ws + 8192 + BUF);
    unsigned short* Bhi = (unsigned short*)(ws + 8192 + 2 * BUF);
    unsigned short* Blo = (unsigned short*)(ws + 8192 + 3 * BUF);
    float* Weff = (float*)(ws + 8192 + 4 * BUF);     // 2*3*256 fp32
    float* beff = Weff + 2 * 3 * HID;                // 6 fp32
    // Pre-swizzled selected-expert weight fragments (max Ksel=32): 8 MB each.
    const size_t WSEL = (size_t)2 * 32 * HID * HID * sizeof(unsigned short);
    unsigned short* Whi = (unsigned short*)(ws + 8192 + 4 * BUF + 8192);
    unsigned short* Wlo = (unsigned short*)(ws + 8192 + 4 * BUF + 8192 + WSEL);

    // 1) top-k gate compaction for all 5 groups x 2 images
    gate_topk<<<dim3(5, 2), 256, 0, stream>>>(raw, selIdx, selVal);

    // 2) layer 0 (E=8, K=4)  -> A buffers
    layer0_kernel<<<NC * HID / 256, 256, 0, stream>>>(coords, W0, b0,
                                                      selIdx, selVal, Ahi, Alo);

    const dim3 grid(NC / 32, HID / 64, 2);
    // 3) layer 1 (E=16, K=4):   A -> B
    prep_weights<<<(2 * 4 * 4096) / 256, 256, 0, stream>>>(W1, selIdx + 8, 4, 16,
                                                           Whi, Wlo);
    moe_layer<<<grid, 256, 0, stream>>>(Whi, Wlo, b1, Ahi, Alo,
                                        selIdx + 8,  selVal + 8,  4,  16,  Bhi, Blo);
    // 4) layer 2 (E=64, K=32):  B -> A
    prep_weights<<<(2 * 32 * 4096) / 256, 256, 0, stream>>>(W2, selIdx + 16, 32, 64,
                                                            Whi, Wlo);
    moe_layer<<<grid, 256, 0, stream>>>(Whi, Wlo, b2, Bhi, Blo,
                                        selIdx + 16, selVal + 16, 32, 64,  Ahi, Alo);
    // 5) layer 3 (E=256, K=32): A -> B
    prep_weights<<<(2 * 32 * 4096) / 256, 256, 0, stream>>>(W3, selIdx + 80, 32, 256,
                                                            Whi, Wlo);
    moe_layer<<<grid, 256, 0, stream>>>(Whi, Wlo, b3, Ahi, Alo,
                                        selIdx + 80, selVal + 80, 32, 256, Bhi, Blo);

    // 6) layer 4 (linear): pre-combine selected experts, then tiny GEMV
    combine4_kernel<<<dim3(3, 2), 256, 0, stream>>>(W4, b4,
                                                    selIdx + 144, selVal + 144,
                                                    Weff, beff);
    final4_kernel<<<2 * NC / 256, 256, 0, stream>>>(Bhi, Blo, Weff, beff,
                                                    (float*)d_out);
}
